// HungarianMatcher_1503238554012
// MI455X (gfx1250) — compile-verified, matrix operand order checked by reading, not run on hardware
//
#include <hip/hip_runtime.h>
#include <hip/hip_bf16.h>

typedef __attribute__((ext_vector_type(16))) _Float16 v16h;
typedef __attribute__((ext_vector_type(8)))  _Float16 v8h;
typedef __attribute__((ext_vector_type(8)))  float    v8f;

#define BSZ 16
#define NQ  900
#define NC  92
#define NCP 96       // K padded to multiple of 32
#define NT_PER 100
#define NT  1600     // BSZ * NT_PER
#define NTT 100      // number of 16-target tiles
#define INF_F 3.0e38f

// ---------------------------------------------------------------------------
// Kernel 1: softmax over 92 classes, one wave (32 lanes) per (b,q) row.
// Writes fp16 probabilities padded to 96 columns into workspace.
// ---------------------------------------------------------------------------
__global__ __launch_bounds__(256) void softmax_probs_kernel(
    const float* __restrict__ logits, _Float16* __restrict__ probs)
{
    const int row  = blockIdx.x * (blockDim.x >> 5) + (threadIdx.x >> 5);
    const int lane = threadIdx.x & 31;
    if (row >= BSZ * NQ) return;

    const float* x = logits + (size_t)row * NC;

    float m = -INF_F;
    for (int c = lane; c < NC; c += 32) m = fmaxf(m, x[c]);
    #pragma unroll
    for (int off = 16; off; off >>= 1) m = fmaxf(m, __shfl_xor(m, off, 32));

    float s = 0.0f;
    for (int c = lane; c < NC; c += 32) s += __expf(x[c] - m);
    #pragma unroll
    for (int off = 16; off; off >>= 1) s += __shfl_xor(s, off, 32);
    const float inv = 1.0f / s;

    _Float16* p = probs + (size_t)row * NCP;
    for (int c = lane; c < NCP; c += 32)
        p[c] = (c < NC) ? (_Float16)(__expf(x[c] - m) * inv) : (_Float16)0.0f;
}

// ---------------------------------------------------------------------------
// Kernel 2: precompute one-hot B fragments, one wave per 16-target tile.
// Fragment stored in per-lane register order: ohbuf[((tt*3+s)*32+lane)*16..]
// so the consumer loads it with two b128 loads per K-step.
// ---------------------------------------------------------------------------
__global__ __launch_bounds__(32) void onehot_frag_kernel(
    const int* __restrict__ tgt_ids, _Float16* __restrict__ ohbuf)
{
    const int tt   = blockIdx.x;
    const int lane = threadIdx.x;
    const int n    = lane & 15;
    const int kofs = (lane >> 4) ? 8 : 0;
    const int cls  = tgt_ids[tt * 16 + n];

    #pragma unroll
    for (int s = 0; s < 3; ++s) {
        v8h lo, hi;
        #pragma unroll
        for (int e = 0; e < 8; ++e) {
            const int k1 = s * 32 + kofs + e;
            lo[e] = (_Float16)((k1      == cls) ? 1.0f : 0.0f);
            hi[e] = (_Float16)((k1 + 16 == cls) ? 1.0f : 0.0f);
        }
        _Float16* dst = ohbuf + (size_t)(((tt * 3 + s) * 32) + lane) * 16;
        *(v8h*)(dst)     = lo;
        *(v8h*)(dst + 8) = hi;
    }
}

// ---------------------------------------------------------------------------
// Kernel 3: each wave computes a 16(q) x 16(t) tile of C.
// cost_class via WMMA: P_tile(16x96 f16) x OneHot(96x16 f16); A and B
// fragments are pure b128 loads. Then L1 + GIoU in VALU, fused store.
// 8 waves/block share the q-tile (A cachelines reused), each owns a t-tile.
// ---------------------------------------------------------------------------
__global__ __launch_bounds__(256) void cost_tile_kernel(
    const _Float16* __restrict__ probs,
    const _Float16* __restrict__ ohbuf,
    const float*    __restrict__ pred_boxes,   // (BSZ*NQ, 4) cxcywh
    const float*    __restrict__ tgt_bbox,     // (NT, 4) xyxy
    float*          __restrict__ C)
{
    const int lane = threadIdx.x & 31;
    const int wv   = threadIdx.x >> 5;
    const int tt   = blockIdx.x * 8 + wv;      // t-tile handled by this wave
    if (tt >= NTT) return;                     // wave-uniform: EXEC stays ~0
    const int b  = blockIdx.z;
    const int q0 = blockIdx.y * 16;
    const int t0 = tt * 16;

    const int mrow = lane & 15;                // A-matrix row this lane supplies
    const int hi   = lane >> 4;                // lane group selects K sub-range
    const int n    = lane & 15;                // B/C/D column for this lane
    const int kofs = hi ? 8 : 0;

    int qa = q0 + mrow; if (qa > NQ - 1) qa = NQ - 1;    // clamp for edge tile
    const _Float16* arow = probs + (size_t)(b * NQ + qa) * NCP + kofs;
    const _Float16* brow = ohbuf + (size_t)((tt * 3) * 32 + lane) * 16;

    v8f acc = {};
    #pragma unroll
    for (int s = 0; s < 3; ++s) {
        const v8h a_lo = *(const v8h*)(arow + s * 32);
        const v8h a_hi = *(const v8h*)(arow + s * 32 + 16);
        const v8h b_lo = *(const v8h*)(brow + s * 32 * 16);
        const v8h b_hi = *(const v8h*)(brow + s * 32 * 16 + 8);
        const v16h a  = __builtin_shufflevector(a_lo, a_hi,
            0,1,2,3,4,5,6,7,8,9,10,11,12,13,14,15);
        const v16h oh = __builtin_shufflevector(b_lo, b_hi,
            0,1,2,3,4,5,6,7,8,9,10,11,12,13,14,15);
        acc = __builtin_amdgcn_wmma_f32_16x16x32_f16(
            false, a, false, oh, (short)0, acc, false, false);
    }
    // acc[r] = prob[q0 + r + 8*hi, cls(t0+n)]

    // target box (xyxy) for this lane's column
    const float4 tb = *(const float4*)(tgt_bbox + (size_t)(t0 + n) * 4);
    const float tcx = 0.5f * (tb.x + tb.z), tcy = 0.5f * (tb.y + tb.w);
    const float tw  = tb.z - tb.x,          th  = tb.w - tb.y;
    const float areaB = tw * th;

    #pragma unroll
    for (int r = 0; r < 8; ++r) {
        const int q = q0 + r + 8 * hi;
        if (q >= NQ) continue;
        const float4 pb = *(const float4*)(pred_boxes + (size_t)(b * NQ + q) * 4);
        const float cx = pb.x, cy = pb.y, w = pb.z, h = pb.w;

        // L1 in cxcywh space
        const float cbbox = fabsf(cx - tcx) + fabsf(cy - tcy)
                          + fabsf(w  - tw ) + fabsf(h  - th );

        // GIoU in xyxy space
        const float px0 = cx - 0.5f * w, py0 = cy - 0.5f * h;
        const float px1 = cx + 0.5f * w, py1 = cy + 0.5f * h;
        const float areaA = (px1 - px0) * (py1 - py0);
        const float ix0 = fmaxf(px0, tb.x), iy0 = fmaxf(py0, tb.y);
        const float ix1 = fminf(px1, tb.z), iy1 = fminf(py1, tb.w);
        const float iw = fmaxf(ix1 - ix0, 0.0f), ih = fmaxf(iy1 - iy0, 0.0f);
        const float inter = iw * ih;
        const float uni   = areaA + areaB - inter;
        const float iou   = inter / uni;
        const float ex0 = fminf(px0, tb.x), ey0 = fminf(py0, tb.y);
        const float ex1 = fmaxf(px1, tb.z), ey1 = fmaxf(py1, tb.w);
        const float cw = fmaxf(ex1 - ex0, 0.0f), ch = fmaxf(ey1 - ey0, 0.0f);
        const float areaC = cw * ch;
        const float giou  = iou - (areaC - uni) / areaC;

        // COST_BBOX*L1 + COST_CLASS*(-prob) + COST_GIOU*(-giou)
        const float cost = 5.0f * cbbox - acc[r] - 2.0f * giou;
        C[((size_t)b * NQ + q) * NT + t0 + n] = cost;
    }
}

// ---------------------------------------------------------------------------
// Kernel 4: Jonker-Volgenant LSA, one 1024-thread block per batch.
// cost(t, q) = C[b, q, b*T + t]; all reads hit the 192MB L2 (C is 92MB).
// ---------------------------------------------------------------------------
__global__ __launch_bounds__(1024) void lsa_kernel(
    const float* __restrict__ C, float* __restrict__ rows_out,
    float* __restrict__ cols_out)
{
    const int b   = blockIdx.x;
    const int tid = threadIdx.x;
    const int NB  = 1024;

    __shared__ float u_sh[NT_PER + 1];
    __shared__ float v_sh[NQ + 1];
    __shared__ float minv[NQ + 1];
    __shared__ int   way[NQ + 1];
    __shared__ int   p_sh[NQ + 1];
    __shared__ int   used[NQ + 1];
    __shared__ float redv[1024];
    __shared__ int   redj[1024];
    __shared__ int   j0_sh, i0_sh, j1_sh;
    __shared__ float delta_sh;

    const float* Cb = C + (size_t)b * NQ * NT + (size_t)b * NT_PER;

    for (int j = tid; j <= NQ; j += NB) { v_sh[j] = 0.0f; p_sh[j] = 0; way[j] = 0; }
    for (int i = tid; i <= NT_PER; i += NB) u_sh[i] = 0.0f;
    __syncthreads();

    for (int i = 1; i <= NT_PER; ++i) {
        for (int j = tid; j <= NQ; j += NB) { minv[j] = INF_F; used[j] = 0; }
        if (tid == 0) { p_sh[0] = i; j0_sh = 0; }
        __syncthreads();

        int done = 0;
        while (!done) {
            if (tid == 0) { used[j0_sh] = 1; i0_sh = p_sh[j0_sh]; }
            __syncthreads();
            const int j0 = j0_sh;
            const int i0 = i0_sh;

            // relax all unused columns against row i0
            const int j = tid + 1;                   // 901 <= 1024: one col/thread
            float cand = INF_F;
            if (j <= NQ && !used[j]) {
                const float cur = Cb[(size_t)(j - 1) * NT + (i0 - 1)]
                                  - u_sh[i0] - v_sh[j];
                if (cur < minv[j]) { minv[j] = cur; way[j] = j0; }
                cand = minv[j];
            }
            redv[tid] = cand;
            redj[tid] = j;
            __syncthreads();

            // arg-min tree reduction, ties -> smallest j (matches np.argmin)
            #pragma unroll
            for (int s = 512; s > 0; s >>= 1) {
                if (tid < s) {
                    const float ov = redv[tid + s];
                    const int   oj = redj[tid + s];
                    if (ov < redv[tid] || (ov == redv[tid] && oj < redj[tid])) {
                        redv[tid] = ov; redj[tid] = oj;
                    }
                }
                __syncthreads();
            }
            if (tid == 0) { delta_sh = redv[0]; j1_sh = redj[0]; }
            __syncthreads();
            const float delta = delta_sh;
            const int   j1    = j1_sh;

            // dual updates
            for (int jj = tid; jj <= NQ; jj += NB) {
                if (used[jj]) { u_sh[p_sh[jj]] += delta; v_sh[jj] -= delta; }
                else          { minv[jj] -= delta; }
            }
            if (tid == 0) j0_sh = j1;
            __syncthreads();
            done = (p_sh[j1] == 0);
        }

        // augment along the alternating path (sequential, thread 0)
        if (tid == 0) {
            int j0 = j0_sh;
            while (j0) {
                const int j1 = way[j0];
                p_sh[j0] = p_sh[j1];
                j0 = j1;
            }
        }
        __syncthreads();
    }

    // emit (rows, cols) sorted by assigned query == argsort(pred_for_tgt)
    if (tid == 0) {
        int cnt = 0;
        for (int j = 1; j <= NQ; ++j) {
            if (p_sh[j] > 0) {
                rows_out[b * NT_PER + cnt] = (float)(j - 1);
                cols_out[b * NT_PER + cnt] = (float)(p_sh[j] - 1);
                ++cnt;
            }
        }
    }
}

// ---------------------------------------------------------------------------
extern "C" void kernel_launch(void* const* d_in, const int* in_sizes, int n_in,
                              void* d_out, int out_size, void* d_ws, size_t ws_size,
                              hipStream_t stream) {
    const float* pred_logits = (const float*)d_in[0];   // (16,900,92)
    const float* pred_boxes  = (const float*)d_in[1];   // (16,900,4)
    const int*   tgt_ids     = (const int*)d_in[2];     // (1600)
    const float* tgt_bbox    = (const float*)d_in[3];   // (1600,4)

    float* C        = (float*)d_out;                    // 16*900*1600
    float* rows_out = C + (size_t)BSZ * NQ * NT;        // +1600
    float* cols_out = rows_out + NT;                    // +1600

    _Float16* probs = (_Float16*)d_ws;                  // 14400*96 fp16 (2.76MB)
    _Float16* ohbuf = probs + (size_t)BSZ * NQ * NCP;   // 100*3*32*16 fp16 (300KB)

    // 1) softmax -> fp16 probs (8 rows per 256-thread block)
    {
        const int rows = BSZ * NQ;
        dim3 grid((rows + 7) / 8), block(256);
        softmax_probs_kernel<<<grid, block, 0, stream>>>(pred_logits, probs);
    }
    // 2) one-hot B fragments, one wave per t-tile
    {
        dim3 grid(NTT), block(32);
        onehot_frag_kernel<<<grid, block, 0, stream>>>(tgt_ids, ohbuf);
    }
    // 3) cost matrix, 16x16 tile per wave (WMMA class-cost + fused VALU geom)
    {
        dim3 grid((NTT + 7) / 8, (NQ + 15) / 16, BSZ), block(256);
        cost_tile_kernel<<<grid, block, 0, stream>>>(
            probs, ohbuf, pred_boxes, tgt_bbox, C);
    }
    // 4) 16 independent JV solvers, one block per batch
    {
        dim3 grid(BSZ), block(1024);
        lsa_kernel<<<grid, block, 0, stream>>>(C, rows_out, cols_out);
    }
}
